// OutputDAG_64579128263358
// MI455X (gfx1250) — matrix operand order repaired and checked by reading, not compiled, not run on hardware
//
#include <hip/hip_runtime.h>
#include <hip/hip_bf16.h>

typedef __attribute__((ext_vector_type(16))) _Float16 v16h;
typedef __attribute__((ext_vector_type(8)))  _Float16 v8h;
typedef __attribute__((ext_vector_type(8)))  float    v8f;

#define NEGV  (-1.0e9f)
#define BB    8
#define VV    512
#define EE    1024
#define HH    16
#define CS    64
#define VOC   32000
#define TT    (BB * VV)      // 4096 rows
#define LP    520            // padded LDS row stride (floats)

// ---------------------------------------------------------------------------
// WMMA fragment loaders (wave32; this code's contiguous k order, same for A and B)
// A 16x32 f16: lane<16 holds K 0..15 contiguous, lane>=16 holds K 16..31
// B 32x16 f16: lane<16 holds K 0..15 contiguous, lane>=16 holds K 16..31
// ---------------------------------------------------------------------------
__device__ __forceinline__ v16h load_a_frag(const _Float16* row_k0, int aoff) {
    const v8h lo = *(const v8h*)(row_k0 + aoff);
    const v8h hi = *(const v8h*)(row_k0 + aoff + 8);
    v16h r;
#pragma unroll
    for (int i = 0; i < 8; ++i) { r[i] = lo[i]; r[i + 8] = hi[i]; }
    return r;
}

// ---------------------------------------------------------------------------
// Kernel: build transposed PHM weight  Wt[n_col][k_row] (f16)
// W[a*kin+k, b*kout+l] = sum_i A[i,a,b]*S[i,k,l];  Wt is N x K row-major.
// ---------------------------------------------------------------------------
__global__ void build_wt_f16_kernel(const float* __restrict__ A,
                                    const float* __restrict__ S,
                                    _Float16* __restrict__ Wt,
                                    int nk, int kin, int kout) {
    size_t K = (size_t)nk * kin;
    size_t N = (size_t)nk * kout;
    size_t idx = (size_t)blockIdx.x * blockDim.x + threadIdx.x;
    if (idx >= N * K) return;
    size_t nIdx = idx / K;
    size_t kIdx = idx - nIdx * K;
    int bb = (int)(nIdx / kout), l = (int)(nIdx % kout);
    int aa = (int)(kIdx / kin),  k = (int)(kIdx % kin);
    float acc = 0.f;
    for (int i = 0; i < nk; ++i)
        acc += A[((size_t)i * nk + aa) * nk + bb] * S[((size_t)i * kin + k) * kout + l];
    Wt[idx] = (_Float16)acc;
}

__global__ void build_wt_f32_kernel(const float* __restrict__ A,
                                    const float* __restrict__ S,
                                    float* __restrict__ Wt,
                                    int nk, int kin, int kout) {
    size_t K = (size_t)nk * kin;
    size_t N = (size_t)nk * kout;
    size_t idx = (size_t)blockIdx.x * blockDim.x + threadIdx.x;
    if (idx >= N * K) return;
    size_t nIdx = idx / K;
    size_t kIdx = idx - nIdx * K;
    int bb = (int)(nIdx / kout), l = (int)(nIdx % kout);
    int aa = (int)(kIdx / kin),  k = (int)(kIdx % kin);
    float acc = 0.f;
    for (int i = 0; i < nk; ++i)
        acc += A[((size_t)i * nk + aa) * nk + bb] * S[((size_t)i * kin + k) * kout + l];
    Wt[idx] = acc;
}

// ---------------------------------------------------------------------------
// Kernel: LayerNorm row -> xn16 (f16), gate logits -> log_softmax -> g (f32)
// one block (256 threads) per row t
// ---------------------------------------------------------------------------
__global__ void ln_gate_kernel(const float* __restrict__ x,
                               const float* __restrict__ ln_g,
                               const float* __restrict__ ln_b,
                               const float* __restrict__ wg,   // [16][1024]
                               const float* __restrict__ gate_b,
                               _Float16* __restrict__ xn16,
                               float* __restrict__ gout) {
    __shared__ float xr[EE];
    __shared__ float red[256];
    __shared__ float sm[18];
    const int t = blockIdx.x, tid = threadIdx.x;
    const float* xrow = x + (size_t)t * EE;

    float s = 0.f, s2 = 0.f;
    for (int k = tid; k < EE; k += 256) {
        float v = xrow[k];
        xr[k] = v; s += v; s2 += v * v;
    }
    red[tid] = s; __syncthreads();
    for (int off = 128; off > 0; off >>= 1) {
        if (tid < off) red[tid] += red[tid + off];
        __syncthreads();
    }
    const float mu = red[0] * (1.0f / EE);
    __syncthreads();
    red[tid] = s2; __syncthreads();
    for (int off = 128; off > 0; off >>= 1) {
        if (tid < off) red[tid] += red[tid + off];
        __syncthreads();
    }
    const float var = red[0] * (1.0f / EE) - mu * mu;
    const float rstd = rsqrtf(var + 1e-5f);
    __syncthreads();

    for (int k = tid; k < EE; k += 256) {
        float v = (xr[k] - mu) * rstd * ln_g[k] + ln_b[k];
        xr[k] = v;
        xn16[(size_t)t * EE + k] = (_Float16)v;
    }
    __syncthreads();

    // gate: 8 waves x 2 heads each, lane-strided dot over 1024
    const int w = tid >> 5, lane = tid & 31;
#pragma unroll
    for (int rep = 0; rep < 2; ++rep) {
        int h = w * 2 + rep;
        float p = 0.f;
        for (int k = lane; k < EE; k += 32) p += xr[k] * wg[(size_t)h * EE + k];
#pragma unroll
        for (int m = 16; m > 0; m >>= 1) p += __shfl_xor(p, m, 32);
        if (lane == 0) sm[h] = p + gate_b[h];
    }
    __syncthreads();
    if (tid == 0) {
        float mx = -3.0e38f;
        for (int h = 0; h < HH; ++h) mx = fmaxf(mx, sm[h]);
        float se = 0.f;
        for (int h = 0; h < HH; ++h) se += __expf(sm[h] - mx);
        sm[16] = mx; sm[17] = __logf(se);
    }
    __syncthreads();
    if (tid < HH) gout[(size_t)t * HH + tid] = sm[tid] - sm[16] - sm[17];
}

// ---------------------------------------------------------------------------
// Kernel: WMMA GEMM  C[T,N] = Am[T,K](f16) * Bt[N,K]^T(f16) + bias
// block = 128 threads (4 waves); each wave: one 16-row tile x 64 cols.
// mode 0: write f32 row-major to Cout
// mode 1: qk epilogue -> q16/k16 packed [B][H][V][64] f16
// ---------------------------------------------------------------------------
__global__ void gemm_wmma_kernel(const _Float16* __restrict__ Am,
                                 const _Float16* __restrict__ Bt,
                                 const float* __restrict__ bias,
                                 int K, int Ntot, int mode,
                                 float* __restrict__ Cout,
                                 _Float16* __restrict__ q16,
                                 _Float16* __restrict__ k16) {
    const int tid = threadIdx.x;
    const int w = tid >> 5, lane = tid & 31;
    const int ln16 = lane & 15, hsel = lane >> 4;
    const int m0 = blockIdx.x * 16;
    const int n0 = blockIdx.y * 256 + w * 64;
    const int aoff = hsel * 16;
    const _Float16* arow = Am + (size_t)(m0 + ln16) * K;
    const _Float16* bbase0 = Bt + (size_t)(n0 + ln16) * K + hsel * 16;

    v8f acc[4] = {v8f{}, v8f{}, v8f{}, v8f{}};
    for (int kk = 0; kk < K; kk += 32) {
        v16h a = load_a_frag(arow + kk, aoff);
        const _Float16* bb = bbase0 + kk;
        v16h b0 = *(const v16h*)(bb);
        v16h b1 = *(const v16h*)(bb + (size_t)16 * K);
        v16h b2 = *(const v16h*)(bb + (size_t)32 * K);
        v16h b3 = *(const v16h*)(bb + (size_t)48 * K);
        acc[0] = __builtin_amdgcn_wmma_f32_16x16x32_f16(false, a, false, b0, (short)0, acc[0], false, false);
        acc[1] = __builtin_amdgcn_wmma_f32_16x16x32_f16(false, a, false, b1, (short)0, acc[1], false, false);
        acc[2] = __builtin_amdgcn_wmma_f32_16x16x32_f16(false, a, false, b2, (short)0, acc[2], false, false);
        acc[3] = __builtin_amdgcn_wmma_f32_16x16x32_f16(false, a, false, b3, (short)0, acc[3], false, false);
    }

#pragma unroll
    for (int t4 = 0; t4 < 4; ++t4) {
        int n = n0 + t4 * 16 + ln16;
        float bv = bias[n];
#pragma unroll
        for (int r = 0; r < 8; ++r) {
            int m = m0 + r + 8 * hsel;
            float v = acc[t4][r] + bv;
            if (mode == 0) {
                Cout[(size_t)m * Ntot + n] = v;
            } else {
                int b = m >> 9, i = m & (VV - 1);
                _Float16 hv = (_Float16)v;
                if (n < EE) {
                    int h = n >> 6, f = n & 63;
                    q16[(((size_t)(b * HH + h)) * VV + i) * CS + f] = hv;
                } else {
                    int nn = n - EE;
                    int h = nn >> 6, f = nn & 63;
                    k16[(((size_t)(b * HH + h)) * VV + i) * CS + f] = hv;
                }
            }
        }
    }
}

// ---------------------------------------------------------------------------
// Kernel: fused scores = q.k^T/8, DAG mask, log_softmax over j, dead-row NEG,
// + gate, online logsumexp over heads -> transition[b][i][j]
// grid = (V/16, B); block = 256 (8 waves); dynamic LDS = 3*16*LP floats
// ---------------------------------------------------------------------------
__global__ void scores_fused_kernel(const _Float16* __restrict__ q16,
                                    const _Float16* __restrict__ k16,
                                    const float* __restrict__ gout,
                                    const int* __restrict__ lens,
                                    float* __restrict__ trans) {
    extern __shared__ float smem[];
    float* sc = smem;
    float* mm = smem + 16 * LP;
    float* ss = smem + 32 * LP;

    const int b  = blockIdx.y;
    const int i0 = blockIdx.x * 16;
    const int len = lens[b];
    const int tid = threadIdx.x;
    const int w = tid >> 5, lane = tid & 31;
    const int ln16 = lane & 15, hsel = lane >> 4;
    const int aoff = hsel * 16;

    for (int h = 0; h < HH; ++h) {
        const _Float16* qrow = q16 + (((size_t)(b * HH + h)) * VV + i0 + ln16) * CS;
        v16h a0 = load_a_frag(qrow, aoff);        // K 0..31
        v16h a1 = load_a_frag(qrow + 32, aoff);   // K 32..63

        for (int jt = w; jt < 32; jt += 8) {
            const int j0 = jt * 16;
            const _Float16* kb = k16 + (((size_t)(b * HH + h)) * VV + j0 + ln16) * CS + hsel * 16;
            v16h b0 = *(const v16h*)(kb);
            v16h b1 = *(const v16h*)(kb + 32);
            v8f acc = v8f{};
            acc = __builtin_amdgcn_wmma_f32_16x16x32_f16(false, a0, false, b0, (short)0, acc, false, false);
            acc = __builtin_amdgcn_wmma_f32_16x16x32_f16(false, a1, false, b1, (short)0, acc, false, false);
            const int j = j0 + ln16;
#pragma unroll
            for (int r = 0; r < 8; ++r) {
                int ri = r + 8 * hsel;
                int i  = i0 + ri;
                bool dead  = (i >= len - 1);
                bool valid = ((j > i) && (j < len)) || (dead && (j == i));
                sc[ri * LP + j] = valid ? acc[r] * 0.125f : NEGV;
            }
        }
        __syncthreads();

        // per-row masked log_softmax over j, then online merge over heads
#pragma unroll
        for (int rep = 0; rep < 2; ++rep) {
            const int ri = w + rep * 8;
            const int i  = i0 + ri;
            float mx = -3.0e38f;
            for (int j = lane; j < VV; j += 32) mx = fmaxf(mx, sc[ri * LP + j]);
#pragma unroll
            for (int m = 16; m > 0; m >>= 1) mx = fmaxf(mx, __shfl_xor(mx, m, 32));
            float se = 0.f;
            for (int j = lane; j < VV; j += 32) se += __expf(sc[ri * LP + j] - mx);
#pragma unroll
            for (int m = 16; m > 0; m >>= 1) se += __shfl_xor(se, m, 32);
            const float lse  = mx + __logf(se);
            const bool dead  = (i >= len - 1);
            const float gh   = gout[((size_t)(b * VV + i)) * HH + h];
            for (int j = lane; j < VV; j += 32) {
                float v = dead ? NEGV : (sc[ri * LP + j] - lse);
                v += gh;
                if (h == 0) {
                    mm[ri * LP + j] = v;
                    ss[ri * LP + j] = 1.0f;
                } else {
                    float mo = mm[ri * LP + j];
                    float nm = fmaxf(mo, v);
                    ss[ri * LP + j] = ss[ri * LP + j] * __expf(mo - nm) + __expf(v - nm);
                    mm[ri * LP + j] = nm;
                }
            }
        }
        __syncthreads();
    }

#pragma unroll
    for (int rep = 0; rep < 2; ++rep) {
        const int ri = w + rep * 8;
        const int i  = i0 + ri;
        for (int j = lane; j < VV; j += 32)
            trans[((size_t)b * VV + i) * VV + j] = mm[ri * LP + j] + __logf(ss[ri * LP + j]);
    }
}

// ---------------------------------------------------------------------------
// Kernel: in-place log_softmax over VOC per row — single online reduction
// pass (running max + scaled sum) then one normalize pass.
// ---------------------------------------------------------------------------
__global__ void vocab_lsm_kernel(float* __restrict__ logits) {
    __shared__ float redm[256];
    __shared__ float reds[256];
    const int t = blockIdx.x, tid = threadIdx.x;
    float* row = logits + (size_t)t * VOC;

    // online (max, sum) in one read pass
    float m = -3.0e38f, s = 0.f;
    for (int j = tid; j < VOC; j += 256) {
        float v = row[j];
        if (v > m) {
            s = s * __expf(m - v) + 1.0f;
            m = v;
        } else {
            s += __expf(v - m);
        }
    }
    redm[tid] = m; reds[tid] = s;
    __syncthreads();
    for (int off = 128; off > 0; off >>= 1) {
        if (tid < off) {
            float m2 = redm[tid + off], s2 = reds[tid + off];
            float m1 = redm[tid],       s1 = reds[tid];
            float nm = fmaxf(m1, m2);
            reds[tid] = s1 * __expf(m1 - nm) + s2 * __expf(m2 - nm);
            redm[tid] = nm;
        }
        __syncthreads();
    }
    const float lse = redm[0] + __logf(reds[0]);
    __syncthreads();
    for (int j = tid; j < VOC; j += 256) row[j] = row[j] - lse;
}

// ---------------------------------------------------------------------------
extern "C" void kernel_launch(void* const* d_in, const int* in_sizes, int n_in,
                              void* d_out, int out_size, void* d_ws, size_t ws_size,
                              hipStream_t stream) {
    const float* x      = (const float*)d_in[0];
    const int*   lens   = (const int*)d_in[1];
    const float* ln_g   = (const float*)d_in[2];
    const float* ln_b   = (const float*)d_in[3];
    const float* attn_A = (const float*)d_in[4];
    const float* attn_S = (const float*)d_in[5];
    const float* attn_b = (const float*)d_in[6];
    const float* gate_A = (const float*)d_in[7];
    const float* gate_S = (const float*)d_in[8];
    const float* gate_b = (const float*)d_in[9];
    const float* lm_A   = (const float*)d_in[10];
    const float* lm_S   = (const float*)d_in[11];
    const float* lm_b   = (const float*)d_in[12];
    float* out = (float*)d_out;

    // workspace carve (all 256B-aligned)
    char* p = (char*)d_ws;
    _Float16* xn16 = (_Float16*)p;  p += (size_t)TT * EE * 2;            // 8 MB
    _Float16* q16  = (_Float16*)p;  p += (size_t)BB * HH * VV * CS * 2;  // 8 MB
    _Float16* k16  = (_Float16*)p;  p += (size_t)BB * HH * VV * CS * 2;  // 8 MB
    _Float16* watt = (_Float16*)p;  p += (size_t)2048 * EE * 2;          // 4 MB
    _Float16* wlm  = (_Float16*)p;  p += (size_t)VOC * EE * 2;           // 62.5 MB
    float*    g32  = (float*)p;     p += (size_t)TT * HH * 4;            // 256 KB
    float*    wg32 = (float*)p;     p += (size_t)HH * EE * 4;            // 64 KB

    float* trans  = out;                           // (B,V,V)
    float* vlogit = out + (size_t)BB * VV * VV;    // (B,V,VOC)

    // 1) build transposed PHM weights
    {
        size_t tot = (size_t)2048 * EE;
        build_wt_f16_kernel<<<dim3((unsigned)((tot + 255) / 256)), 256, 0, stream>>>(
            attn_A, attn_S, watt, 4, 256, 512);
    }
    {
        size_t tot = (size_t)VOC * EE;
        build_wt_f16_kernel<<<dim3((unsigned)((tot + 255) / 256)), 256, 0, stream>>>(
            lm_A, lm_S, wlm, 2, 512, 16000);
    }
    {
        size_t tot = (size_t)HH * EE;
        build_wt_f32_kernel<<<dim3((unsigned)((tot + 255) / 256)), 256, 0, stream>>>(
            gate_A, gate_S, wg32, 4, 256, 4);
    }

    // 2) layernorm + gate log_softmax
    ln_gate_kernel<<<dim3(TT), 256, 0, stream>>>(x, ln_g, ln_b, wg32, gate_b, xn16, g32);

    // 3) qk = xn @ W_attn (+b) -> q16/k16 packed [B][H][V][64]
    gemm_wmma_kernel<<<dim3(TT / 16, 2048 / 256), 128, 0, stream>>>(
        xn16, watt, attn_b, EE, 2048, 1, (float*)nullptr, q16, k16);

    // 4) lm logits = xn @ W_lm (+b) -> d_out vocab region (f32)
    gemm_wmma_kernel<<<dim3(TT / 16, VOC / 256), 128, 0, stream>>>(
        xn16, wlm, lm_b, EE, VOC, 0, vlogit, (_Float16*)nullptr, (_Float16*)nullptr);

    // 5) fused scores / mask / log_softmax_j / +gate / logsumexp_h -> transition
    {
        size_t smem = (size_t)3 * 16 * LP * sizeof(float);  // ~100 KB dynamic LDS
        scores_fused_kernel<<<dim3(VV / 16, BB), 256, smem, stream>>>(
            q16, k16, g32, lens, trans);
    }

    // 6) vocab log_softmax in place
    vocab_lsm_kernel<<<dim3(TT), 256, 0, stream>>>(vlogit);
}
